// Attention_24464133718579
// MI455X (gfx1250) — matrix operand order repaired and checked by reading, not compiled, not run on hardware
//
#include <hip/hip_runtime.h>
#include <hip/hip_bf16.h>
#include <math.h>

// ---------------------------------------------------------------------------
// CDNA5 (gfx1250) attention stack.
// Big GEMMs: v_wmma_f32_16x16x32_bf16 (wave32 WMMA), fp32 accumulate.
// bf16 operands end-to-end; GEMM hot loop is a double-buffered LDS pipeline:
// global_load_b128 for step i+1 issued before the WMMAs of step i, so the
// s_wait_loadcnt lands after the matrix math.  LDS rows padded to 80B for
// conflict-free ds_load_b128 fragment loads.
// ---------------------------------------------------------------------------

typedef __attribute__((ext_vector_type(16))) __bf16 v16bf;
typedef __attribute__((ext_vector_type(8)))  float  v8f;
typedef unsigned short ushort_t;

__device__ __forceinline__ ushort_t f2bf(float f) {
  unsigned int u = __float_as_uint(f);
  unsigned int r = u + 0x7FFFu + ((u >> 16) & 1u);   // round-to-nearest-even
  return (ushort_t)(r >> 16);
}
__device__ __forceinline__ float bflo(unsigned int u) {
  return __uint_as_float(u << 16);
}
__device__ __forceinline__ float bfhi(unsigned int u) {
  return __uint_as_float(u & 0xFFFF0000u);
}

// ------------------------- tiled WMMA GEMM ---------------------------------
// C[M,N] = act(A[M,K] * W[K,N] + bias[N]);  A bf16 [M][K], Wt bf16 [N][K].
// 256 threads = 8 waves; wave tile = 32 x (16*TN) (2 x TN WMMA tiles).
// BM x BN block tile, K stepped by 32, double-buffered LDS.
template<int BM, int BN, int WN, int TN>
__global__ __launch_bounds__(256, 1)
void gemm_bf16_kernel(const ushort_t* __restrict__ A,
                      const ushort_t* __restrict__ Wt,
                      const float* __restrict__ bias,
                      float* __restrict__ Cf, ushort_t* __restrict__ Cb,
                      int M, int N, int K, int act)
{
  constexpr int LDT  = 40;                 // padded LDS row stride (ushorts, 80B)
  constexpr int ASEG = (BM * 4 + 255) / 256;
  constexpr int BSEG = (BN * 4) / 256;

  __shared__ __align__(16) ushort_t lA[2][BM * LDT];
  __shared__ __align__(16) ushort_t lB[2][BN * LDT];

  const int tid   = threadIdx.x;
  const int lane  = tid & 31;
  const int wave  = tid >> 5;
  const int waveM = wave / WN;
  const int waveN = wave % WN;
  const int m0    = blockIdx.x * BM;
  const int n0    = blockIdx.y * BN;
  const int hi    = lane >> 4;
  const int ml    = lane & 15;

  // ---- per-thread staging descriptors (hoisted; advanced +64B per K-step) --
  const uint4* aSrc[ASEG]; int aOff[ASEG]; bool aVal[ASEG];
  #pragma unroll
  for (int j = 0; j < ASEG; ++j) {
    int s = tid + j * 256;
    aVal[j] = (s < BM * 4);
    int r  = (s >> 2) & (BM - 1);
    int sg = s & 3;
    aSrc[j] = (const uint4*)(A + (size_t)(m0 + r) * K) + sg;
    aOff[j] = r * LDT + sg * 8;
  }
  const uint4* bSrc[BSEG]; int bOff[BSEG];
  #pragma unroll
  for (int j = 0; j < BSEG; ++j) {
    int s = tid + j * 256;
    int r = s >> 2, sg = s & 3;
    bSrc[j] = (const uint4*)(Wt + (size_t)(n0 + r) * K) + sg;
    bOff[j] = r * LDT + sg * 8;
  }

  uint4 aReg[ASEG], bReg[BSEG];
  auto loadRegs = [&]() {
    #pragma unroll
    for (int j = 0; j < ASEG; ++j)
      if (aVal[j]) {
        aReg[j] = *aSrc[j];
        __builtin_prefetch((const void*)(aSrc[j] + 4), 0, 1);
        aSrc[j] += 4;
      }
    #pragma unroll
    for (int j = 0; j < BSEG; ++j) {
      bReg[j] = *bSrc[j];
      __builtin_prefetch((const void*)(bSrc[j] + 4), 0, 1);
      bSrc[j] += 4;
    }
  };
  auto storeRegs = [&](int buf) {
    #pragma unroll
    for (int j = 0; j < ASEG; ++j)
      if (aVal[j]) *(uint4*)&lA[buf][aOff[j]] = aReg[j];
    #pragma unroll
    for (int j = 0; j < BSEG; ++j)
      *(uint4*)&lB[buf][bOff[j]] = bReg[j];
  };

  const v8f vzero = {0.f,0.f,0.f,0.f,0.f,0.f,0.f,0.f};
  v8f acc[2][TN];
  for (int i = 0; i < 2; ++i)
    for (int j = 0; j < TN; ++j) acc[i][j] = vzero;

  union Frag { unsigned int u[8]; v16bf v; };

  auto compute = [&](int buf) {
    // fragments (ISA 7.12.2 16-bit layouts), conflict-free b128 LDS loads
    Frag afrag[2], bfrag[TN];
    #pragma unroll
    for (int tm = 0; tm < 2; ++tm) {
      int row = waveM * 32 + tm * 16 + ml;
      const uint4* rowp = (const uint4*)&lA[buf][row * LDT];
      uint4 lo = rowp[hi];        // K = hi*8    .. +7  -> vgprs 0..3
      uint4 hh = rowp[2 + hi];    // K = 16+hi*8 .. +7  -> vgprs 4..7
      afrag[tm].u[0] = lo.x; afrag[tm].u[1] = lo.y;
      afrag[tm].u[2] = lo.z; afrag[tm].u[3] = lo.w;
      afrag[tm].u[4] = hh.x; afrag[tm].u[5] = hh.y;
      afrag[tm].u[6] = hh.z; afrag[tm].u[7] = hh.w;
    }
    #pragma unroll
    for (int tn = 0; tn < TN; ++tn) {
      int col = waveN * (16 * TN) + tn * 16 + ml;
      const uint4* colp = (const uint4*)&lB[buf][col * LDT];
      uint4 lo = colp[hi], hh = colp[2 + hi];
      bfrag[tn].u[0] = lo.x; bfrag[tn].u[1] = lo.y;
      bfrag[tn].u[2] = lo.z; bfrag[tn].u[3] = lo.w;
      bfrag[tn].u[4] = hh.x; bfrag[tn].u[5] = hh.y;
      bfrag[tn].u[6] = hh.z; bfrag[tn].u[7] = hh.w;
    }
    #pragma unroll
    for (int tm = 0; tm < 2; ++tm)
      #pragma unroll
      for (int tn = 0; tn < TN; ++tn)
        acc[tm][tn] = __builtin_amdgcn_wmma_f32_16x16x32_bf16(
            false, afrag[tm].v, false, bfrag[tn].v,
            (short)0, acc[tm][tn], false, false);
  };

  // ---- double-buffered pipeline ----
  const int nsteps = K / 32;
  loadRegs();
  storeRegs(0);
  for (int i = 0; i < nsteps; ++i) {
    __syncthreads();                       // buf[i&1] fully staged
    const bool more = (i + 1 < nsteps);
    if (more) loadRegs();                  // global loads for i+1 in flight
    compute(i & 1);                        // WMMAs hide the load latency
    __syncthreads();                       // all reads of buf[i&1] done
    if (more) storeRegs((i + 1) & 1);      // s_wait_loadcnt lands here
  }

  // ---- epilogue: bias + activation, fp32 and/or bf16 stores ----
  #pragma unroll
  for (int tm = 0; tm < 2; ++tm) {
    #pragma unroll
    for (int tn = 0; tn < TN; ++tn) {
      int nn = n0 + waveN * (16 * TN) + tn * 16 + ml;
      float bv = bias ? bias[nn] : 0.0f;
      #pragma unroll
      for (int r = 0; r < 8; ++r) {
        int mm = m0 + waveM * 32 + tm * 16 + hi * 8 + r;
        float c = acc[tm][tn][r] + bv;
        if (act) c = (c > 0.f) ? c : 0.01f * c;      // leaky_relu
        if (Cf) Cf[(size_t)mm * N + nn] = c;
        if (Cb) Cb[(size_t)mm * N + nn] = f2bf(c);
      }
    }
  }
}

// ------------------------- fp32 -> bf16 elementwise ------------------------
__global__ __launch_bounds__(256)
void cvt_bf16_kernel(const float* __restrict__ src, ushort_t* __restrict__ dst, int n)
{
  int i = (blockIdx.x * 256 + threadIdx.x) * 8;
  if (i + 8 <= n) {
    const float4* s = (const float4*)(src + i);
    float4 a = s[0], b = s[1];
    ushort_t r[8] = { f2bf(a.x), f2bf(a.y), f2bf(a.z), f2bf(a.w),
                      f2bf(b.x), f2bf(b.y), f2bf(b.z), f2bf(b.w) };
    *(uint4*)(dst + i) = *(const uint4*)r;
  }
}

// ------------------- weight transpose-convert: [K][N]f32 -> [N][K]bf16 -----
__global__ __launch_bounds__(256)
void wt_kernel(const float* __restrict__ W, ushort_t* __restrict__ Wt, int K, int N)
{
  __shared__ ushort_t t[32][33];
  const int n0 = blockIdx.x * 32, k0 = blockIdx.y * 32;
  const int c = threadIdx.x & 31, r0 = threadIdx.x >> 5;   // 8 rows / iter
  for (int r = r0; r < 32; r += 8)
    t[c][r] = f2bf(W[(size_t)(k0 + r) * N + n0 + c]);      // coalesced read
  __syncthreads();
  for (int r = r0; r < 32; r += 8)
    Wt[(size_t)(n0 + r) * K + k0 + c] = t[r][c];           // coalesced write
}

// ------------------------- attention (per b,h) -----------------------------
__global__ __launch_bounds__(256)
void attn_kernel(const float* __restrict__ q, const ushort_t* __restrict__ Kb,
                 const ushort_t* __restrict__ Vb, const float* __restrict__ start,
                 const float* __restrict__ endi, const float* __restrict__ temp,
                 ushort_t* __restrict__ vals, float* __restrict__ wbuf)
{
  const int b = blockIdx.x, h = blockIdx.y, tid = threadIdx.x;
  __shared__ float qs[64];
  __shared__ float sc[512];
  __shared__ float red[256];
  __shared__ float pv[4][64];

  if (tid < 64) qs[tid] = q[b * 1024 + h * 64 + tid];
  __syncthreads();

  const float s = floorf(start[b]);
  const float e = ceilf(endi[b]);
  const bool allm = (e < s) || (e < 0.f) || (s > 511.f);
  const float scale = temp[0] * 0.125f;                    // 1/sqrt(DK=64)

  for (int t = tid; t < 512; t += 256) {
    const uint4* kp = (const uint4*)(Kb + (size_t)(b * 512 + t) * 1024 + h * 64);
    float d = 0.f;
    #pragma unroll
    for (int i = 0; i < 8; ++i) {
      uint4 p = kp[i];
      d += qs[i*8+0] * bflo(p.x) + qs[i*8+1] * bfhi(p.x)
         + qs[i*8+2] * bflo(p.y) + qs[i*8+3] * bfhi(p.y)
         + qs[i*8+4] * bflo(p.z) + qs[i*8+5] * bfhi(p.z)
         + qs[i*8+6] * bflo(p.w) + qs[i*8+7] * bfhi(p.w);
    }
    float ft = (float)t;
    bool masked = (ft < s) || (ft > e);
    sc[t] = allm ? 1.0f : (masked ? -INFINITY : d * scale);
  }
  __syncthreads();

  float lm = fmaxf(sc[tid], sc[tid + 256]);
  red[tid] = lm; __syncthreads();
  for (int o = 128; o > 0; o >>= 1) {
    if (tid < o) red[tid] = fmaxf(red[tid], red[tid + o]);
    __syncthreads();
  }
  const float mx = red[0]; __syncthreads();

  float ls = 0.f;
  for (int t = tid; t < 512; t += 256) {
    float w = __expf(sc[t] - mx);
    sc[t] = w; ls += w;
  }
  red[tid] = ls; __syncthreads();
  for (int o = 128; o > 0; o >>= 1) {
    if (tid < o) red[tid] += red[tid + o];
    __syncthreads();
  }
  const float inv = 1.0f / red[0]; __syncthreads();

  for (int t = tid; t < 512; t += 256) {
    float w = sc[t] * inv;
    sc[t] = w;
    wbuf[(size_t)(b * 16 + h) * 512 + t] = w;
  }
  __syncthreads();

  const int d = tid & 63, ch = tid >> 6;
  float p = 0.f;
  for (int t = ch * 128; t < ch * 128 + 128; ++t)
    p += sc[t] * bflo((unsigned int)Vb[(size_t)(b * 512 + t) * 1024 + h * 64 + d]);
  pv[ch][d] = p;
  __syncthreads();
  if (ch == 0)
    vals[b * 1024 + h * 64 + d] = f2bf(pv[0][d] + pv[1][d] + pv[2][d] + pv[3][d]);
}

// ------------------------- layernorm (optional residual, dual output) ------
__global__ __launch_bounds__(256)
void ln_kernel(const float* __restrict__ x, const float* __restrict__ add,
               float* __restrict__ outF, ushort_t* __restrict__ outB)
{
  const int b = blockIdx.x, tid = threadIdx.x;
  __shared__ float row[1024];
  __shared__ float red[256];

  float l = 0.f;
  for (int i = tid; i < 1024; i += 256) {
    float v = x[b * 1024 + i];
    if (add) v += add[b * 1024 + i];
    row[i] = v; l += v;
  }
  red[tid] = l; __syncthreads();
  for (int o = 128; o > 0; o >>= 1) {
    if (tid < o) red[tid] += red[tid + o];
    __syncthreads();
  }
  const float mean = red[0] * (1.f / 1024.f); __syncthreads();

  float lv = 0.f;
  for (int i = tid; i < 1024; i += 256) { float d = row[i] - mean; lv += d * d; }
  red[tid] = lv; __syncthreads();
  for (int o = 128; o > 0; o >>= 1) {
    if (tid < o) red[tid] += red[tid + o];
    __syncthreads();
  }
  const float rstd = rsqrtf(red[0] * (1.f / 1024.f) + 1e-5f);
  __syncthreads();
  for (int i = tid; i < 1024; i += 256) {
    float v = (row[i] - mean) * rstd;
    if (outF) outF[b * 1024 + i] = v;
    if (outB) outB[b * 1024 + i] = f2bf(v);
  }
}

// ------------------------- misc small kernels ------------------------------
__global__ void concat_bf_kernel(const float* __restrict__ a, const float* __restrict__ b2,
                                 ushort_t* __restrict__ o)
{
  int i = blockIdx.x * 256 + threadIdx.x;     // 32*2048 total
  if (i < 32 * 2048) {
    int r = i >> 11, c = i & 2047;
    o[i] = f2bf((c < 1024) ? a[r * 1024 + c] : b2[r * 1024 + (c - 1024)]);
  }
}

__global__ void attmean_kernel(const float* __restrict__ wbuf, float* __restrict__ att)
{
  int b = blockIdx.x, tid = threadIdx.x;
  for (int t = tid; t < 512; t += 256) {
    float s = 0.f;
    for (int h = 0; h < 16; ++h) s += wbuf[(size_t)(b * 16 + h) * 512 + t];
    att[b * 512 + t] = s * (1.f / 16.f);
  }
}

// ---------------------------------------------------------------------------
extern "C" void kernel_launch(void* const* d_in, const int* in_sizes, int n_in,
                              void* d_out, int out_size, void* d_ws, size_t ws_size,
                              hipStream_t stream)
{
  (void)in_sizes; (void)n_in; (void)out_size; (void)ws_size;

  // ---- inputs (setup_inputs() dict order, recursive insertion order) ----
  const float* enc_seq   = (const float*)d_in[0];   // [32,512,1024]
  const float* enc_key   = (const float*)d_in[1];   // [32,512,1024]
  const float* e_l       = (const float*)d_in[2];
  const float* e_r       = (const float*)d_in[3];
  const float* start_ind = (const float*)d_in[4];
  const float* end_ind   = (const float*)d_in[5];

  const float *qnW[5], *qnB[5];
  for (int i = 0; i < 5; ++i) {
    qnW[i] = (const float*)d_in[6 + 2 * i];
    qnB[i] = (const float*)d_in[7 + 2 * i];
  }
  struct Layer { const float *qW,*qb,*kW,*kb,*vW,*vb,*oW,*ob,*pW[4],*pb[4],*temp; } L[3];
  for (int l = 0; l < 3; ++l) {
    int base = 16 + 17 * l;
    L[l].qW = (const float*)d_in[base + 0];  L[l].qb = (const float*)d_in[base + 1];
    L[l].kW = (const float*)d_in[base + 2];  L[l].kb = (const float*)d_in[base + 3];
    L[l].vW = (const float*)d_in[base + 4];  L[l].vb = (const float*)d_in[base + 5];
    L[l].oW = (const float*)d_in[base + 6];  L[l].ob = (const float*)d_in[base + 7];
    for (int p = 0; p < 4; ++p) {
      L[l].pW[p] = (const float*)d_in[base + 8 + 2 * p];
      L[l].pb[p] = (const float*)d_in[base + 9 + 2 * p];
    }
    L[l].temp = (const float*)d_in[base + 16];
  }
  const float* foW = (const float*)d_in[67];
  const float* fob = (const float*)d_in[68];

  // ---- workspace carve-up ----
  char* p = (char*)d_ws;
  auto carve = [&](size_t bytes) -> void* {
    void* r = (void*)p; p += (bytes + 255) & ~(size_t)255; return r;
  };
  float*    query_f = (float*)carve(32 * 1024 * 4);
  float*    rawb_f  = (float*)carve(32 * 1024 * 4);
  float*    h2f     = (float*)carve(32 * 1024 * 4);
  float*    qbuf    = (float*)carve(32 * 1024 * 4);
  float*    wbuf    = (float*)carve(32 * 16 * 512 * 4);
  ushort_t* cat_b   = (ushort_t*)carve(32 * 2048 * 2);
  ushort_t* h1b     = (ushort_t*)carve(32 * 1024 * 2);
  ushort_t* h2b     = (ushort_t*)carve(32 * 1024 * 2);
  ushort_t* query_b = (ushort_t*)carve(32 * 1024 * 2);
  ushort_t* xbuf_b  = (ushort_t*)carve(32 * 1024 * 2);
  ushort_t* vals_b  = (ushort_t*)carve(32 * 1024 * 2);
  ushort_t* rawb_b  = (ushort_t*)carve(32 * 1024 * 2);
  ushort_t* enc_key_b = (ushort_t*)carve((size_t)16384 * 1024 * 2);
  ushort_t* enc_seq_b = (ushort_t*)carve((size_t)16384 * 1024 * 2);
  ushort_t* kbb     = (ushort_t*)carve((size_t)16384 * 1024 * 2);
  ushort_t* vbb     = (ushort_t*)carve((size_t)16384 * 1024 * 2);
  ushort_t* wts     = (ushort_t*)carve((size_t)2048 * 1024 * 2);  // reused Wt scratch

  // GEMM with on-the-fly weight transpose-convert (stream serializes kernels,
  // so a single Wt scratch is reused for every GEMM).
  auto gemmT = [&](const ushort_t* A, const float* W, const float* bias,
                   float* Cf, ushort_t* Cb, int M, int N, int K, int act) {
    wt_kernel<<<dim3(N / 32, K / 32), 256, 0, stream>>>(W, wts, K, N);
    if (M == 32) {
      gemm_bf16_kernel<32, 128, 8, 1><<<dim3(1, N / 128), 256, 0, stream>>>(
          A, wts, bias, Cf, Cb, M, N, K, act);
    } else {
      gemm_bf16_kernel<128, 64, 2, 2><<<dim3(M / 128, N / 64), 256, 0, stream>>>(
          A, wts, bias, Cf, Cb, M, N, K, act);
    }
  };

  // ---- pre-convert encoder streams (reused by all 3 layers) ----
  const int ENC = 16384 * 1024;
  cvt_bf16_kernel<<<ENC / 2048, 256, 0, stream>>>(enc_key, enc_key_b, ENC);
  cvt_bf16_kernel<<<ENC / 2048, 256, 0, stream>>>(enc_seq, enc_seq_b, ENC);

  // ---- query_net over concat(e_l, e_r) ----
  concat_bf_kernel<<<(32 * 2048 + 255) / 256, 256, 0, stream>>>(e_l, e_r, cat_b);
  gemmT(cat_b, qnW[0], qnB[0], nullptr, h1b,    32, 1024, 2048, 1);
  gemmT(h1b,   qnW[1], qnB[1], nullptr, h2b,    32, 1024, 1024, 1);
  gemmT(h2b,   qnW[2], qnB[2], nullptr, h1b,    32, 1024, 1024, 1);
  gemmT(h1b,   qnW[3], qnB[3], nullptr, h2b,    32, 1024, 1024, 1);
  gemmT(h2b,   qnW[4], qnB[4], query_f, query_b, 32, 1024, 1024, 0);

  // ---- attention layers ----
  for (int l = 0; l < 3; ++l) {
    gemmT(query_b,   L[l].qW, L[l].qb, qbuf,  nullptr, 32,    1024, 1024, 0);
    gemmT(enc_key_b, L[l].kW, L[l].kb, nullptr, kbb,   16384, 1024, 1024, 0);
    gemmT(enc_seq_b, L[l].vW, L[l].vb, nullptr, vbb,   16384, 1024, 1024, 0);
    attn_kernel<<<dim3(32, 16), 256, 0, stream>>>(qbuf, kbb, vbb, start_ind, end_ind,
                                                  L[l].temp, vals_b, wbuf);
    gemmT(vals_b, L[l].oW, L[l].ob, rawb_f, rawb_b, 32, 1024, 1024, 0);
    ln_kernel<<<32, 256, 0, stream>>>(rawb_f, nullptr, nullptr, xbuf_b);
    gemmT(xbuf_b, L[l].pW[0], L[l].pb[0], nullptr, h1b, 32, 1024, 1024, 1);
    gemmT(h1b,    L[l].pW[1], L[l].pb[1], nullptr, h2b, 32, 1024, 1024, 1);
    gemmT(h2b,    L[l].pW[2], L[l].pb[2], nullptr, h1b, 32, 1024, 1024, 1);
    gemmT(h1b,    L[l].pW[3], L[l].pb[3], h2f, nullptr, 32, 1024, 1024, 0);
    ln_kernel<<<32, 256, 0, stream>>>(h2f, query_f, query_f, query_b);
  }

  // ---- outputs ----
  gemmT(rawb_b, foW, fob, (float*)d_out, nullptr, 32, 1024, 1024, 0);
  attmean_kernel<<<32, 256, 0, stream>>>(wbuf, (float*)d_out + 32 * 1024);
}
